// LiquidReservoir_40793599378017
// MI455X (gfx1250) — matrix operand order repaired and checked by reading, not compiled, not run on hardware
//
#include <hip/hip_runtime.h>
#include <math.h>

// Problem constants (from the reference)
#define T_STEPS 2048
#define BATCH   64
#define IN_SZ   80    // padded to 96 (3 k-tiles of 32)
#define INPAD   96
#define RES     500   // padded to 512 (32 n-tiles of 16, 16 k-tiles of 32)
#define RPAD    512
#define KLO     8     // k-tiles 0..7  (K < 256): W_res B-frags from LDS
#define KHI     8     // k-tiles 8..15 (K >= 256): W_res B-frags in VGPRs
#define LEAK    0.3f

typedef __bf16 v16bf __attribute__((ext_vector_type(16)));
typedef float  v8f   __attribute__((ext_vector_type(8)));

union FragBF { unsigned u[8]; uint4 q[2]; __bf16 h[16]; v16bf v; };

#if __has_builtin(__builtin_amdgcn_tanhf)
#define TANH_F32(v) __builtin_amdgcn_tanhf(v)
#else
#define TANH_F32(v) tanhf(v)
#endif

// 4 independent workgroups (one per 16-row batch tile; batch rows are
// independent in the recurrence -> zero cross-WG sync). 512 threads =
// 16 wave32 waves; wave w owns output-column tiles {w, w+16}.
//
// Register-pressure plan (allocator ceiling observed ~256 VGPRs/wave):
//   persistent = W_res upper-K half (128) + W_input (48) = 176 VGPRs
//   W_res lower-K half (K<256, all 512 cols) lives in LDS: 256 KB bf16
//   state (32 KB) + x tile (6 KB) double-buffered in LDS -> 294 KB total
// All LDS fragment gathers are 16B-aligned uint4 reads -> ds_load_b128.
// fp32 master state stays in registers; one s_barrier per step.
__global__ __launch_bounds__(512)
void reservoir_fused(const float* __restrict__ x,        // [T, B, IN]
                     const float* __restrict__ s0,       // [B, RES]
                     const float* __restrict__ Wres,     // [RES, RES]
                     const float* __restrict__ Winp,     // [RES, IN]
                     float* __restrict__ out)            // [T*B*RES] ++ [B*RES]
{
    extern __shared__ char smem[];
    unsigned* sst = (unsigned*)smem;                   // state: [2][16][RPAD/2], 32 KB
    unsigned* xS  = (unsigned*)(smem + 32768);         // x tile: [2][16][INPAD/2], 6 KB
    unsigned* wrS = (unsigned*)(smem + 32768 + 6144);  // W_res K<256: [512][128] dw, 256 KB

    const int tid    = threadIdx.x;
    const int lane   = tid & 31;
    const int wave   = tid >> 5;        // 0..15
    const int halfHi = lane >> 4;       // 0: lanes 0-15, 1: lanes 16-31
    const int ln     = lane & 15;
    const int b0     = blockIdx.x * 16; // batch tile base
    const int abase  = halfHi * 8;      // accumulator row base: M = i + 8*half
    const int bbase  = halfHi * 16;     // B K-base: contiguous k = 16h + e
    const int rA     = wave * 16 + ln;        // tile-A column (< 256, never padded)
    const int rBcol  = 256 + wave * 16 + ln;  // tile-B column (< 512, pad >= 500)

    // ---- Stage W_res lower-K half into LDS: bf16 [n(512)][k(256)] --------
    __bf16* wrSh = (__bf16*)wrS;
    for (int i = tid; i < RPAD * 256; i += 512) {
        const int n = i >> 8, k = i & 255;            // k < 256 < RES always
        wrSh[i] = (n < RES) ? (__bf16)Wres[n * RES + k] : (__bf16)0.0f;
    }
    // ---- Stage x[0] tile into LDS buffer 0 (bf16, zero-padded to 96) -----
    __bf16* xSh = (__bf16*)xS;
#pragma unroll
    for (int s = 0; s < 3; ++s) {
        const int ii = tid + s * 512;
        const int m2 = ii / INPAD, k2 = ii % INPAD;
        xSh[ii] = (k2 < IN_SZ) ? (__bf16)x[(b0 + m2) * IN_SZ + k2] : (__bf16)0.0f;
    }

    // ---- Persistent VGPR fragments ---------------------------------------
    // B[k][n] = W_res[n][k]; upper-K half only (K = 256 + kk*32 + ...)
    v16bf wfA[KHI], wfB[KHI];
#pragma unroll
    for (int kk = 0; kk < KHI; ++kk) {
        FragBF fa, fb;
#pragma unroll
        for (int e = 0; e < 16; ++e) {
            const int k = 256 + kk * 32 + bbase + e;   // guard folds for kk<7
            fa.h[e] = (k < RES) ? (__bf16)Wres[rA * RES + k] : (__bf16)0.0f;
            const float vb = (k < RES && rBcol < RES) ? Wres[rBcol * RES + k] : 0.0f;
            fb.h[e] = (__bf16)vb;
        }
        wfA[kk] = fa.v;
        wfB[kk] = fb.v;
    }
    // W_input B-fragments (K padded 80 -> 96)
    v16bf wiA[3], wiB[3];
#pragma unroll
    for (int kk = 0; kk < 3; ++kk) {
        FragBF fa, fb;
#pragma unroll
        for (int e = 0; e < 16; ++e) {
            const int k = kk * 32 + bbase + e;
            fa.h[e] = (k < IN_SZ) ? (__bf16)Winp[rA * IN_SZ + k] : (__bf16)0.0f;
            const float vb = (k < IN_SZ && rBcol < RES) ? Winp[rBcol * IN_SZ + k] : 0.0f;
            fb.h[e] = (__bf16)vb;
        }
        wiA[kk] = fa.v;
        wiB[kk] = fb.v;
    }

    // ---- fp32 master state in registers; bf16 copy into LDS buf 0 --------
    // C layout: VGPR i, lane L: M = i + (L<16?0:8), N = L%16.
    float sFa[8], sFb[8];
    __bf16* st0 = (__bf16*)sst;
#pragma unroll
    for (int i = 0; i < 8; ++i) {
        const int m = i + abase;
        sFa[i] = s0[(b0 + m) * RES + rA];
        sFb[i] = (rBcol < RES) ? s0[(b0 + m) * RES + rBcol] : 0.0f;
        st0[m * RPAD + rA]    = (__bf16)sFa[i];
        st0[m * RPAD + rBcol] = (__bf16)sFb[i];   // zeros in padding columns
    }
    __syncthreads();

    // ---- Hoisted per-thread gather bases (all uint4 / 16B aligned) -------
    const int mrow = ln;                               // A-matrix row for this lane
    const int adw  = bbase >> 1;                       // A gather dword base {0,8}
    const unsigned* stRow = sst + mrow * (RPAD / 2) + adw;
    const unsigned* xRow  = xS  + mrow * (INPAD / 2) + adw;
    const unsigned* wrA0  = wrS + rA    * 128 + (bbase >> 1);  // lower-K B base
    const unsigned* wrB0  = wrS + rBcol * 128 + (bbase >> 1);
    // staging indices (fixed per thread across all steps)
    int stg_idx[3], stg_off[3]; bool stg_ok[3];
#pragma unroll
    for (int s = 0; s < 3; ++s) {
        const int ii = tid + s * 512;
        stg_idx[s] = ii;
        const int m2 = ii / INPAD, k2 = ii % INPAD;
        stg_off[s] = (b0 + m2) * IN_SZ + k2;
        stg_ok[s]  = (k2 < IN_SZ);
    }

#define LOAD_A(frag, base, kk)                                            \
    do {                                                                  \
        (frag).q[0] = *(const uint4*)((base) + (kk) * 16);                \
        (frag).q[1] = *(const uint4*)((base) + (kk) * 16 + 4);            \
    } while (0)
#define LOAD_B(frag, base, kk)                                            \
    do {                                                                  \
        (frag).q[0] = *(const uint4*)((base) + (kk) * 16);                \
        (frag).q[1] = *(const uint4*)((base) + (kk) * 16 + 4);            \
    } while (0)
#define WMMA(c, a, b) \
    __builtin_amdgcn_wmma_f32_16x16x32_bf16(false, (a), false, (b), (short)0, (c), false, false)

    // ---- Sequential scan over time ---------------------------------------
    for (int t = 0; t < T_STEPS; ++t) {
        const int buf = t & 1;
        if (tid < 16 && t + 2 < T_STEPS)   // warm x two steps ahead
            __builtin_prefetch(x + (size_t)(t + 2) * BATCH * IN_SZ + (b0 + tid) * IN_SZ, 0, 3);

        v8f c0 = {}, c1 = {};   // two independent accumulator chains

        // drive: c += x[t, tile, :] @ W_input^T  (A from LDS x-tile, B in regs)
        const unsigned* xr = xRow + buf * (16 * (INPAD / 2));
#pragma unroll
        for (int kk = 0; kk < 3; ++kk) {
            FragBF a;
            LOAD_A(a, xr, kk);                 // zeros live in LDS padding
            c0 = WMMA(c0, a.v, wiA[kk]);
            c1 = WMMA(c1, a.v, wiB[kk]);
        }

        // issue next step's x loads early; consumed after the recurrent chain
        float xstage[3];
        const bool do_stage = (t + 1 < T_STEPS);
        if (do_stage) {
            const float* xnext = x + (size_t)(t + 1) * BATCH * IN_SZ;
#pragma unroll
            for (int s = 0; s < 3; ++s)
                xstage[s] = stg_ok[s] ? xnext[stg_off[s]] : 0.0f;
        }

        const unsigned* sb = stRow + buf * (16 * (RPAD / 2));
        // recurrent, K < 256: A and B both gathered from LDS (b128 reads)
#pragma unroll
        for (int kk = 0; kk < KLO; ++kk) {
            FragBF a, ba, bb;
            LOAD_A(a,  sb,   kk);
            LOAD_B(ba, wrA0, kk);
            LOAD_B(bb, wrB0, kk);
            c0 = WMMA(c0, a.v, ba.v);
            c1 = WMMA(c1, a.v, bb.v);
        }
        // recurrent, K >= 256: B persistent in VGPRs; ping-pong A fragments
        FragBF aE, aO;
        LOAD_A(aE, sb, KLO);
#pragma unroll
        for (int kp = 0; kp < KHI / 2; ++kp) {
            LOAD_A(aO, sb, KLO + 2 * kp + 1);
            c0 = WMMA(c0, aE.v, wfA[2 * kp]);
            c1 = WMMA(c1, aE.v, wfB[2 * kp]);
            if (kp < KHI / 2 - 1) LOAD_A(aE, sb, KLO + 2 * kp + 2);
            c0 = WMMA(c0, aO.v, wfA[2 * kp + 1]);
            c1 = WMMA(c1, aO.v, wfB[2 * kp + 1]);
        }

        // pointwise: tanh + leaky blend; stream out; bf16 state -> other buffer
        float* orow = out + (size_t)t * BATCH * RES;
        __bf16* dst = (__bf16*)(sst + (buf ^ 1) * (16 * (RPAD / 2)));
#pragma unroll
        for (int i = 0; i < 8; ++i) {
            const int m = i + abase;
            const float nwa = (1.0f - LEAK) * sFa[i] + LEAK * TANH_F32(c0[i]);
            const float nwb = (1.0f - LEAK) * sFb[i] + LEAK * TANH_F32(c1[i]);
            sFa[i] = nwa;
            sFb[i] = nwb;                       // stays 0 for padded columns
            __builtin_nontemporal_store(nwa, orow + (b0 + m) * RES + rA);
            if (rBcol < RES)
                __builtin_nontemporal_store(nwb, orow + (b0 + m) * RES + rBcol);
            dst[m * RPAD + rA]    = (__bf16)nwa;
            dst[m * RPAD + rBcol] = (__bf16)nwb;
        }
        // commit next x tile to the other LDS buffer
        if (do_stage) {
            __bf16* xd = xSh + (buf ^ 1) * (16 * INPAD);
#pragma unroll
            for (int s = 0; s < 3; ++s)
                xd[stg_idx[s]] = (__bf16)xstage[s];
        }
        __syncthreads();   // next step reads the ^1 buffers written above
    }

    // ---- final_state, concatenated after all_states ----------------------
    float* fin = out + (size_t)T_STEPS * BATCH * RES;
#pragma unroll
    for (int i = 0; i < 8; ++i) {
        const int m = i + abase;
        fin[(b0 + m) * RES + rA] = sFa[i];
        if (rBcol < RES) fin[(b0 + m) * RES + rBcol] = sFb[i];
    }

#undef LOAD_A
#undef LOAD_B
#undef WMMA
}

extern "C" void kernel_launch(void* const* d_in, const int* in_sizes, int n_in,
                              void* d_out, int out_size, void* d_ws, size_t ws_size,
                              hipStream_t stream) {
    const float* x    = (const float*)d_in[0];  // [2048, 64, 80]
    const float* s0   = (const float*)d_in[1];  // [64, 500]
    const float* wres = (const float*)d_in[2];  // [500, 500]
    const float* winp = (const float*)d_in[3];  // [500, 80]
    float* out = (float*)d_out;                 // [2048*64*500 + 64*500]
    (void)in_sizes; (void)n_in; (void)out_size; (void)d_ws; (void)ws_size;

    const size_t lds_bytes = 32768        // state double buffer
                           + 6144         // x tile double buffer
                           + 262144;      // W_res lower-K half (bf16)
    dim3 grid(BATCH / 16);   // 4 independent batch-tile workgroups
    dim3 block(512);         // 16 wave32 waves, 2 n-tiles each
    reservoir_fused<<<grid, block, lds_bytes, stream>>>(x, s0, wres, winp, out);
}
